// NARX_32186484916903
// MI455X (gfx1250) — compile-verified, matrix operand order checked by reading, not run on hardware
//
#include <hip/hip_runtime.h>

// ---------------- NARX autoregressive forward, MI455X (gfx1250) ----------
// B=256, T=2048, d_x=32, d_y=16, d_hl=512, D_I=D_O=8, d_in=384, steps=2040.
// 16 persistent workgroups (16 batch rows each = one WMMA M-tile), 8 wave32.
// Recurrence is independent per batch row -> no inter-block sync; only 3
// in-WGP barriers per step. W1 fragments are loop-invariant and get hoisted
// into VGPRs by the compiler (register-resident weights).

#define TT    2048
#define DX    32
#define DY    16
#define DH    512
#define DIN   384
#define STEPS 2040
#define NT1   32      // GEMM1 N-tiles (512/16)
#define KT1   12      // GEMM1 K-tiles (384/32)
#define KT2   16      // GEMM2 K-tiles (512/32)

typedef __attribute__((ext_vector_type(16))) __bf16 v16bf;
typedef __attribute__((ext_vector_type(8)))  float  v8f;

__device__ __forceinline__ unsigned short f2bfu(float f) {
  unsigned u = __builtin_bit_cast(unsigned, f);
  u += 0x7FFFu + ((u >> 16) & 1u);          // round-to-nearest-even
  return (unsigned short)(u >> 16);
}

__device__ __forceinline__ v8f vzero8() {
  v8f z = {0.f,0.f,0.f,0.f,0.f,0.f,0.f,0.f};
  return z;
}

// CDNA5 native TRANS tanh; v_nop covers the TRANS result-use hazard since
// the compiler cannot see through the asm block.
__device__ __forceinline__ float fast_tanh(float x) {
  float r;
  asm volatile("v_tanh_f32 %0, %1\n\tv_nop" : "=v"(r) : "v"(x));
  return r;
}

// bf16 fragment element K-index for 16x32 A / 32x16 B tiles (ISA 7.12.2):
// lane l: row/col = l&15, khalf = l>>4; element e (0..15):
//   K = (e&8)*2 + khalf*8 + (e&7)
// Note: even/odd element pairs (e, e+1) map to adjacent K -> u32 moves.
__device__ __forceinline__ int frag_k(int e, int khalf) {
  return ((e & 8) << 1) + khalf * 8 + (e & 7);
}

// ---------------- one-shot weight repack: f32 -> bf16 fragment-major -----
__global__ __launch_bounds__(256) void narx_pack_weights(
    const float* __restrict__ W1, const float* __restrict__ W2,
    unsigned short* __restrict__ wsW1, unsigned short* __restrict__ wsW2) {
  int gid = blockIdx.x * 256 + threadIdx.x;
  const int nW1 = NT1 * KT1 * 512;
  if (gid < nW1) {
    int frag = gid >> 9, idx = gid & 511;
    int lane = idx >> 4, e = idx & 15;
    int ntile = frag / KT1, ktile = frag - ntile * KT1;
    int n = ntile * 16 + (lane & 15);
    int k = ktile * 32 + frag_k(e, lane >> 4);
    wsW1[gid] = f2bfu(W1[k * DH + n]);
  } else {
    int g = gid - nW1;
    if (g < KT2 * 512) {
      int frag = g >> 9, idx = g & 511;
      int lane = idx >> 4, e = idx & 15;
      int n = lane & 15;
      int k = frag * 32 + frag_k(e, lane >> 4);
      wsW2[g] = f2bfu(W2[k * DY + n]);
    }
  }
}

// build the x-window A-fragment for time t into ring slot t&7.
// 256 threads x 1 pair: float2 global load -> packed u32 bf16 LDS store.
__device__ __forceinline__ void build_xfrag(unsigned short* xfrag,
                                            const float* __restrict__ x,
                                            int rowbase, int t, int tid) {
  int slot = t & 7;
  int f = tid * 2;                      // even element index, 0..510
  int lane2 = f >> 4, e = f & 15;
  int m = lane2 & 15;
  int kk = frag_k(e, lane2 >> 4);       // column within d_x (==32), even
  const float* p = &x[((size_t)(rowbase + m) * TT + t) * DX + kk];
  float x0 = p[0], x1 = p[1];
  unsigned v = (unsigned)f2bfu(x0) | ((unsigned)f2bfu(x1) << 16);
  *(unsigned*)(xfrag + slot * 512 + f) = v;
}

// ---------------- persistent recurrence kernel ---------------------------
__global__ __launch_bounds__(256) void narx_forward(
    const float* __restrict__ x, const float* __restrict__ b1,
    const float* __restrict__ b2,
    const unsigned short* __restrict__ wsW1,
    const unsigned short* __restrict__ wsW2,
    float* __restrict__ y) {
  __shared__ __attribute__((aligned(32))) unsigned short xfrag[8 * 512];  // x-window A-frags (ring by t&7)
  __shared__ __attribute__((aligned(32))) unsigned short cfrag[4 * 512];  // carry A-frags (lag-major)
  __shared__ __attribute__((aligned(32))) unsigned short hfrag[16 * 512]; // h as GEMM2 A-frags
  __shared__ __attribute__((aligned(32))) unsigned short ringB[8 * 256];  // bf16 feedback ring [slot][m][dy]
  __shared__ __attribute__((aligned(32))) float pbuf[8 * 256];            // GEMM2 partials [wave][lane][8]

  const int tid = threadIdx.x;
  const int lane = tid & 31;
  const int wave = tid >> 5;
  const int rowbase = blockIdx.x * 16;

  // biases into registers
  float b1r[4];
#pragma unroll
  for (int j = 0; j < 4; ++j) b1r[j] = b1[(wave * 4 + j) * 16 + (lane & 15)];
  const float b2r = b2[lane & 15];

  // init: zero ring + carry frags (y-hat(0..7)=0), zero first 8 timesteps of y
  for (int f = tid; f < 1024; f += 256) {
    ((unsigned*)ringB)[f] = 0u;
    ((unsigned*)cfrag)[f] = 0u;
  }
  for (int f = tid; f < 16 * 8 * 16; f += 256) {
    int m = f >> 7, rem = f & 127;
    int t = rem >> 4, n = rem & 15;
    y[((size_t)(rowbase + m) * TT + t) * DY + n] = 0.f;
  }
  // prebuild x fragments for t = 0..7 (t = i+8 is built inside the loop)
  for (int t = 0; t < 8; ++t) build_xfrag(xfrag, x, rowbase, t, tid);
  __syncthreads();

  for (int i = 0; i < STEPS; ++i) {
    // ---- phase 1: GEMM1  h = tanh(inp @ W1 + b1), per-wave 4 N-tiles ----
    v8f acc[4];
#pragma unroll
    for (int j = 0; j < 4; ++j) acc[j] = vzero8();
#pragma unroll
    for (int kt = 0; kt < KT1; ++kt) {
      v16bf a;
      if (kt < 8)
        a = *(const v16bf*)(xfrag + ((i + kt) & 7) * 512 + lane * 16);
      else
        a = *(const v16bf*)(cfrag + (kt - 8) * 512 + lane * 16);
#pragma unroll
      for (int j = 0; j < 4; ++j) {
        int frag = (wave * 4 + j) * KT1 + kt;   // loop-invariant -> VGPR-resident
        v16bf w = *(const v16bf*)(wsW1 + frag * 512 + lane * 16);
        acc[j] = __builtin_amdgcn_wmma_f32_16x16x32_bf16(
            false, a, false, w, (short)0, acc[j], false, false);
      }
    }
    // bias + tanh, scatter into GEMM2 A-fragment layout in LDS
#pragma unroll
    for (int j = 0; j < 4; ++j) {
      int kh = (wave * 4 + j) * 16 + (lane & 15);  // h column == GEMM2 K index
      int kt = kh >> 5, kk = kh & 31;
      int khalf = (kk >> 3) & 1;
      int e2 = ((kk >> 4) << 3) | (kk & 7);
#pragma unroll
      for (int e = 0; e < 8; ++e) {
        float v = fast_tanh(acc[j][e] + b1r[j]);
        int m = e + (lane >> 4) * 8;
        hfrag[kt * 512 + (m | (khalf << 4)) * 16 + e2] = f2bfu(v);
      }
    }
    __syncthreads();

    // ---- phase 2: GEMM2 partials; overlap: stage x(t=i+8) and the lag<7
    //      part of next step's carry fragments (u32 bf16 moves, no converts)
    build_xfrag(xfrag, x, rowbase, i + 8, tid);
    for (int fp = tid; fp < 1024; fp += 256) {
      int f = fp * 2;
      int pfrag = f >> 9, idx = f & 511;
      int lane2 = idx >> 4, e = idx & 15;
      int m = lane2 & 15;
      int kk = frag_k(e, lane2 >> 4);            // even
      int lag = pfrag * 2 + (kk >> 4);
      if (lag < 7) {                             // lag 7 written by wave 0 below
        int dy = kk & 15;
        unsigned v = *(const unsigned*)(ringB +
                     ((i + 1 + lag) & 7) * 256 + m * 16 + dy);
        *(unsigned*)(cfrag + f) = v;
      }
    }
    {
      v8f p = vzero8();
#pragma unroll
      for (int q = 0; q < 2; ++q) {
        int kt = wave * 2 + q;
        v16bf a = *(const v16bf*)(hfrag + kt * 512 + lane * 16);
        v16bf w = *(const v16bf*)(wsW2 + kt * 512 + lane * 16);
        p = __builtin_amdgcn_wmma_f32_16x16x32_bf16(
            false, a, false, w, (short)0, p, false, false);
      }
      *(v8f*)(pbuf + (wave * 32 + lane) * 8) = p;
    }
    __syncthreads();

    // ---- phase 3: wave 0 reduces partials, adds b2; writes y, the bf16
    //      ring slot, and the lag-7 quarter of next step's carry fragment ----
    if (wave == 0) {
      v8f s = *(const v8f*)(pbuf + lane * 8);
#pragma unroll
      for (int w2 = 1; w2 < 8; ++w2) {
        v8f p = *(const v8f*)(pbuf + (w2 * 32 + lane) * 8);
        s = s + p;
      }
      const int n = lane & 15;
      const int t = i + 8;
      const int rs = (t & 7) * 256;
      const int khalf7 = (n >> 3) & 1;           // lag 7 -> kk = 16+n
      const int e7 = 8 | (n & 7);
#pragma unroll
      for (int e = 0; e < 8; ++e) {
        float v = s[e] + b2r;
        int m = e + (lane >> 4) * 8;
        unsigned short hb = f2bfu(v);
        ringB[rs + m * 16 + n] = hb;                               // ring
        cfrag[3 * 512 + (m | (khalf7 << 4)) * 16 + e7] = hb;       // lag-7
        y[((size_t)(rowbase + m) * TT + t) * DY + n] = v;          // output
      }
    }
    __syncthreads();
  }
}

extern "C" void kernel_launch(void* const* d_in, const int* in_sizes, int n_in,
                              void* d_out, int out_size, void* d_ws,
                              size_t ws_size, hipStream_t stream) {
  const float* x  = (const float*)d_in[0];
  const float* W1 = (const float*)d_in[1];
  const float* b1 = (const float*)d_in[2];
  const float* W2 = (const float*)d_in[3];
  const float* b2 = (const float*)d_in[4];
  float* y = (float*)d_out;

  unsigned short* wsW1 = (unsigned short*)d_ws;            // 384 frags * 1KB
  unsigned short* wsW2 = wsW1 + NT1 * KT1 * 512;           // 16 frags * 1KB
  const int totalPack = NT1 * KT1 * 512 + KT2 * 512;       // 204800 elems

  narx_pack_weights<<<(totalPack + 255) / 256, 256, 0, stream>>>(W1, W2,
                                                                 wsW1, wsW2);
  narx_forward<<<16, 256, 0, stream>>>(x, b1, b2, wsW1, wsW2, y);
}
